// BianGuaAttention_87153476370484
// MI455X (gfx1250) — matrix operand order repaired and checked by reading, not compiled, not run on hardware
//
#include <hip/hip_runtime.h>
#include <hip/hip_bf16.h>
#include <stdint.h>

// Problem constants (match reference)
#define TSEQ  2048
#define NB    2
#define NH    12
#define DH    64
#define CM    768
#define MROWS (NB*TSEQ)     // 4096 token rows
#define KT32  (CM/32)       // 24 K-steps of 32
#define NT16  (CM/16)       // 48 N-tiles of 16

typedef __attribute__((ext_vector_type(16))) __bf16 v16bf;
typedef __attribute__((ext_vector_type(8)))  float  v8f;
typedef __attribute__((ext_vector_type(4)))  unsigned int u32x4;
typedef __attribute__((ext_vector_type(8)))  int i32x8;
typedef __attribute__((ext_vector_type(4)))  int i32x4;

union Frag { v16bf v; uint4 q[2]; };

__device__ __forceinline__ unsigned short f2bf(float f) {
  union { float f; unsigned u; } x; x.f = f;
  unsigned u = x.u;
  u += 0x7FFFu + ((u >> 16) & 1u);   // round-to-nearest-even
  return (unsigned short)(u >> 16);
}

__device__ __forceinline__ v8f wmma_bf16(const Frag& a, const Frag& b, v8f c) {
  return __builtin_amdgcn_wmma_f32_16x16x32_bf16(
      /*neg_a=*/false, a.v, /*neg_b=*/false, b.v,
      /*c_mod=*/(short)0, c, /*reuse_a=*/false, /*reuse_b=*/false);
}

// ---- Tensor Data Mover: copy one packed weight panel (4 fragments of 1KB,
// stride 24KB) into LDS.  2D descriptor per CDNA5 ISA.
__device__ __forceinline__ void tdm_load_panel(const unsigned short* gsrc,
                                               unsigned int lds_off) {
  unsigned long long ga = (unsigned long long)(uintptr_t)gsrc;
  u32x4 g0;
  g0[0] = 1u;                                   // count=1, user mode
  g0[1] = lds_off;                              // lds_addr (bytes)
  g0[2] = (unsigned int)(ga & 0xFFFFFFFFu);     // global_addr[31:0]
  g0[3] = (unsigned int)((ga >> 32) & 0x01FFFFFFu) | (2u << 30); // ga[56:32] | type=2
  i32x8 g1;
  g1[0] = (int)(2u << 16);     // workgroup_mask=0, data_size=2 (4 bytes)
  g1[1] = (int)(256u << 16);   // atomic_bar_addr=0 | tensor_dim0[15:0]=256
  g1[2] = (int)(4u << 16);     // tensor_dim0[31:16]=0 | tensor_dim1[15:0]=4
  g1[3] = (int)(256u << 16);   // tensor_dim1[31:16]=0 | tile_dim0=256 dwords
  g1[4] = 4;                   // tile_dim1=4, tile_dim2=0
  g1[5] = 6144;                // tensor_dim0_stride[31:0] = 6144 dwords (24KB)
  g1[6] = 0;                   // stride hi | tensor_dim1_stride lo
  g1[7] = 0;
  i32x4 z4 = {0, 0, 0, 0};
#if defined(__clang_major__) && (__clang_major__ >= 23)
  i32x8 z8 = {0, 0, 0, 0, 0, 0, 0, 0};
  __builtin_amdgcn_tensor_load_to_lds(g0, g1, z4, z4, z8, 0);
#else
  __builtin_amdgcn_tensor_load_to_lds(g0, g1, z4, z4, 0);
#endif
}

// ---------------------------------------------------------------- prep ----
__global__ void cvt_bf16_kernel(const float* __restrict__ in,
                                unsigned short* __restrict__ out, int n) {
  int i = blockIdx.x * blockDim.x + threadIdx.x;
  if (i < n) out[i] = f2bf(in[i]);
}

// soft_hex[bt][c] = sum_j hex_weights[bt][j] * hexagrams[j][c]
__global__ void soft_hex_kernel(const float* __restrict__ hw,
                                const float* __restrict__ hex,
                                float* __restrict__ sh) {
  int i = blockIdx.x * blockDim.x + threadIdx.x;
  if (i >= MROWS * 6) return;
  int bt = i / 6, c = i % 6;
  float s = 0.0f;
  #pragma unroll
  for (int j = 0; j < 64; ++j) s += hw[(size_t)bt * 64 + j] * hex[j * 6 + c];
  sh[i] = s;
}

// Pack W [768x768] fp32 into WMMA B-fragment layout (bf16):
// fragment (nt,kt): lane = N column nt*16+lane%16, 16 halfs = K in table order (8h+j, 16+8h+j-8)
__global__ void pack_weight_kernel(const float* __restrict__ W,
                                   unsigned short* __restrict__ WB) {
  int lin = blockIdx.x * blockDim.x + threadIdx.x;
  const int total = NT16 * KT32 * 32;
  if (lin >= total) return;
  int lane = lin & 31;
  int tile = lin >> 5;
  int kt = tile % KT32;
  int nt = tile / KT32;
  int n = nt * 16 + (lane & 15);
  int kb0 = kt * 32 + (lane >> 4) * 8;
  unsigned short* out = WB + (size_t)lin * 16;
  #pragma unroll
  for (int j = 0; j < 16; ++j)
    out[j] = f2bf(W[(size_t)n * CM + kb0 + (j < 8 ? j : j + 8)]);   // B[k][n] = W[n][k]
}

// ---------------------------------------------------------------- GEMM ----
// Y[M,768] = A_bf16[M,768] @ W^T via packed WB fragments.
// Block = 4 waves, 64 output rows, one head (64-wide N block).  The weight
// panel for each k-step (4 fragments = 4KB) is TDM-copied into LDS, double
// buffered with s_wait_tensorcnt; all 4 waves consume it via ds_load_b128.
// mode 0: store bf16 [B,H,T,D] (Q) | 1: bf16 [B,H,D,T] (K^T) |
// mode 2: store bf16 [B,H,T,D] (V) | 3: store fp32 [B*T,768] (final out)
__global__ __launch_bounds__(128)
void gemm_bf16_wmma(const unsigned short* __restrict__ A,
                    const unsigned short* __restrict__ WB,
                    unsigned short* __restrict__ outH,
                    float* __restrict__ outF, int mode) {
  const int h    = blockIdx.y;            // 64-wide N block == one head
  const int wid  = threadIdx.x >> 5;
  const int lane = threadIdx.x & 31;
  const int m0   = blockIdx.x * 64 + wid * 16;
  const int nk   = lane & 15;
  const int hlf  = lane >> 4;
  const int rbase = hlf * 8;              // C-layout row base for this lane
  const int base0 = hlf * 8;              // A-layout K base for this lane

  __shared__ __align__(16) unsigned short ldsB[2][4 * 32 * 16]; // 2 x 4KB

  const unsigned short* wpanel = WB + (size_t)(4 * h) * KT32 * 512;

  if (wid == 0)
    tdm_load_panel(wpanel, (unsigned int)(uintptr_t)&ldsB[0][0]);

  v8f acc[4] = {};
  const unsigned short* arow = A + (size_t)(m0 + nk) * CM;

  for (int kt = 0; kt < KT32; ++kt) {
    const int cur = kt & 1;
    if (wid == 0) {
      if (kt + 1 < KT32) {
        tdm_load_panel(wpanel + (size_t)(kt + 1) * 512,
                       (unsigned int)(uintptr_t)&ldsB[cur ^ 1][0]);
        __builtin_amdgcn_s_wait_tensorcnt(1);   // oldest (panel kt) complete
      } else {
        __builtin_amdgcn_s_wait_tensorcnt(0);
      }
    }
    __syncthreads();                            // panel kt visible to all waves

    Frag a;
    const unsigned short* ap = arow + kt * 32 + base0;
    a.q[0] = *(const uint4*)(ap);               // K = base0 .. base0+7
    a.q[1] = *(const uint4*)(ap + 16);          // K = base0+16 .. base0+23
    #pragma unroll
    for (int f = 0; f < 4; ++f) {
      Frag b;
      const unsigned short* bp = &ldsB[cur][(f * 32 + lane) * 16];
      b.q[0] = ((const uint4*)bp)[0];
      b.q[1] = ((const uint4*)bp)[1];
      acc[f] = wmma_bf16(a, b, acc[f]);
    }
    __syncthreads();                            // all waves done with panel kt
  }

  #pragma unroll
  for (int f = 0; f < 4; ++f) {
    #pragma unroll
    for (int r = 0; r < 8; ++r) {
      const int row = m0 + rbase + r;
      const float val = acc[f][r];
      if (mode == 3) {
        outF[(size_t)row * CM + h * DH + f * 16 + nk] = val;
      } else {
        const int b = row / TSEQ, t = row % TSEQ;
        const unsigned short hv = f2bf(val);
        if (mode == 1)
          outH[((size_t)(b * NH + h) * DH + f * 16 + nk) * TSEQ + t] = hv;
        else
          outH[((size_t)(b * NH + h) * TSEQ + t) * DH + f * 16 + nk] = hv;
      }
    }
  }
}

// ------------------------------------------------------ flash attention ----
__global__ __launch_bounds__(32)
void flash_attn_wmma(const unsigned short* __restrict__ qbf,   // [B,H,T,D]
                     const unsigned short* __restrict__ ktbf,  // [B,H,D,T]
                     const unsigned short* __restrict__ vprj,  // [B,H,T,D]
                     const float* __restrict__ shx,            // [B*T,6]
                     const float* __restrict__ lamp,
                     unsigned short* __restrict__ attn) {      // [B*T,768] bf16
  const int q0   = blockIdx.x * 16;
  const int h    = blockIdx.y;
  const int b    = blockIdx.z;
  const int lane = threadIdx.x;
  const int nk   = lane & 15;
  const int hlf  = lane >> 4;
  const int rbase = hlf * 8;
  const int base0 = hlf * 8;
  const int bh = b * NH + h;

  __shared__ __align__(16) unsigned short Plds[16 * 32];
  __shared__ float shq[16][6];

  // stage 16 query soft_hex rows (96 floats, 3 per lane)
  #pragma unroll
  for (int j = 0; j < 3; ++j) {
    int i = lane * 3 + j;
    shq[i / 6][i % 6] = shx[(size_t)(b * TSEQ + q0 + i / 6) * 6 + (i % 6)];
  }
  __syncthreads();

  float sig;
  { float lam = lamp[0]; sig = 1.0f / (1.0f + __expf(-lam)); }
  const float hb = 0.5f * sig;
  const float scale = 0.125f;             // 64^-0.5

  // preload Q A-fragments (K = D = 64 split into two k-steps)
  Frag qa[2];
  {
    const unsigned short* qrow = qbf + ((size_t)bh * TSEQ + q0 + nk) * DH;
    #pragma unroll
    for (int c = 0; c < 2; ++c) {
      const unsigned short* p = qrow + c * 32 + base0;
      qa[c].q[0] = *(const uint4*)p;
      qa[c].q[1] = *(const uint4*)(p + 16);
    }
  }

  v8f O[4] = {};
  float mrow[8], lrow[8];
  #pragma unroll
  for (int r = 0; r < 8; ++r) { mrow[r] = -1e30f; lrow[r] = 0.0f; }

  const int nChunks = (q0 + 16 + 31) / 32;
  for (int cc = 0; cc < nChunks; ++cc) {
    const int s0 = cc * 32;

    // ---- scores: S = Q @ K^T for 32 keys (two 16-key C tiles)
    v8f Sa = {}, Sb = {};
    #pragma unroll
    for (int c = 0; c < 2; ++c) {
      const unsigned short* kr =
          ktbf + ((size_t)bh * DH + c * 32 + base0) * TSEQ + s0 + nk;
      Frag kb;
      Frag kb2;
      #pragma unroll
      for (int j = 0; j < 16; ++j) {
        const size_t ko = (size_t)(j < 8 ? j : j + 8) * TSEQ;
        ((unsigned short*)&kb)[j]  = kr[ko];
        ((unsigned short*)&kb2)[j] = kr[ko + 16];
      }
      Sa = wmma_bf16(qa[c], kb, Sa);
      Sb = wmma_bf16(qa[c], kb2, Sb);
    }

    // per-lane key soft_hex vectors for this lane's two key columns
    float ska[6], skb[6];
    {
      const float* pA = shx + (size_t)(b * TSEQ + s0 + nk) * 6;
      const float* pB = shx + (size_t)(b * TSEQ + s0 + 16 + nk) * 6;
      #pragma unroll
      for (int c2 = 0; c2 < 6; ++c2) { ska[c2] = pA[c2]; skb[c2] = pB[c2]; }
    }

    // ---- bias + mask + online softmax (half-wave shuffles per row)
    #pragma unroll
    for (int r = 0; r < 8; ++r) {
      const int row = rbase + r;
      const int qi = q0 + row;
      float da = 0.0f, db = 0.0f;
      #pragma unroll
      for (int c2 = 0; c2 < 6; ++c2) {
        da += shq[row][c2] * ska[c2];
        db += shq[row][c2] * skb[c2];
      }
      float sa = Sa[r] * scale + hb * da;
      float sb = Sb[r] * scale + hb * db;
      if (s0 + nk > qi)      sa = -1e30f;
      if (s0 + 16 + nk > qi) sb = -1e30f;
      float rm = fmaxf(sa, sb);
      #pragma unroll
      for (int msk = 1; msk < 16; msk <<= 1) rm = fmaxf(rm, __shfl_xor(rm, msk, 32));
      const float mnew  = fmaxf(mrow[r], rm);
      const float alpha = __expf(mrow[r] - mnew);
      const float ea = __expf(sa - mnew);
      const float eb = __expf(sb - mnew);
      float rs = ea + eb;
      #pragma unroll
      for (int msk = 1; msk < 16; msk <<= 1) rs += __shfl_xor(rs, msk, 32);
      lrow[r] = lrow[r] * alpha + rs;
      mrow[r] = mnew;
      O[0][r] *= alpha; O[1][r] *= alpha; O[2][r] *= alpha; O[3][r] *= alpha;
      Plds[row * 32 + nk]      = f2bf(ea);
      Plds[row * 32 + 16 + nk] = f2bf(eb);
    }
    __syncthreads();

    // ---- P (C-layout) -> A-fragment via LDS, then O += P @ V
    Frag pf;
    {
      const unsigned short* pp = &Plds[nk * 32 + base0];
      pf.q[0] = *(const uint4*)pp;
      pf.q[1] = *(const uint4*)(pp + 16);
    }
    #pragma unroll
    for (int f = 0; f < 4; ++f) {
      const unsigned short* vr =
          vprj + ((size_t)bh * TSEQ + s0 + base0) * DH + f * 16 + nk;
      Frag vb;
      #pragma unroll
      for (int j = 0; j < 16; ++j)
        ((unsigned short*)&vb)[j] = vr[(size_t)(j < 8 ? j : j + 8) * DH];
      O[f] = wmma_bf16(pf, vb, O[f]);
    }
    __syncthreads();
  }

  // ---- epilogue: normalize rows, store bf16 [B,T,H*D]
  #pragma unroll
  for (int r = 0; r < 8; ++r) {
    const int row = rbase + r;
    const float inv = 1.0f / lrow[r];
    const size_t obase = ((size_t)(b * TSEQ) + q0 + row) * CM + h * DH;
    #pragma unroll
    for (int f = 0; f < 4; ++f)
      attn[obase + f * 16 + nk] = f2bf(O[f][r] * inv);
  }
}

// -------------------------------------------------------------- launch ----
extern "C" void kernel_launch(void* const* d_in, const int* in_sizes, int n_in,
                              void* d_out, int out_size, void* d_ws, size_t ws_size,
                              hipStream_t stream) {
  (void)in_sizes; (void)n_in; (void)out_size; (void)ws_size;
  const float* x   = (const float*)d_in[0];
  const float* hw  = (const float*)d_in[1];
  const float* Wq  = (const float*)d_in[2];
  const float* Wk  = (const float*)d_in[3];
  const float* Wv  = (const float*)d_in[4];
  const float* Wo  = (const float*)d_in[5];
  const float* lam = (const float*)d_in[6];
  const float* hex = (const float*)d_in[7];
  float* out = (float*)d_out;

  char* ws = (char*)d_ws;
  size_t off = 0;
  auto alloc = [&](size_t bytes) -> void* {
    void* p = ws + off; off += (bytes + 255) & ~(size_t)255; return p;
  };
  unsigned short* xbf    = (unsigned short*)alloc((size_t)MROWS * CM * 2);
  unsigned short* WBq    = (unsigned short*)alloc((size_t)CM * CM * 2);
  unsigned short* WBk    = (unsigned short*)alloc((size_t)CM * CM * 2);
  unsigned short* WBv    = (unsigned short*)alloc((size_t)CM * CM * 2);
  unsigned short* WBo    = (unsigned short*)alloc((size_t)CM * CM * 2);
  float*          shx    = (float*)alloc((size_t)MROWS * 6 * 4);
  unsigned short* qbf    = (unsigned short*)alloc((size_t)MROWS * CM * 2);
  unsigned short* ktbf   = (unsigned short*)alloc((size_t)MROWS * CM * 2);
  unsigned short* vprj   = (unsigned short*)alloc((size_t)MROWS * CM * 2);
  unsigned short* attnbf = (unsigned short*)alloc((size_t)MROWS * CM * 2);

  const int nx = MROWS * CM;
  cvt_bf16_kernel<<<(nx + 255) / 256, 256, 0, stream>>>(x, xbf, nx);
  soft_hex_kernel<<<(MROWS * 6 + 255) / 256, 256, 0, stream>>>(hw, hex, shx);

  const int wpack = NT16 * KT32 * 32;
  pack_weight_kernel<<<(wpack + 255) / 256, 256, 0, stream>>>(Wq, WBq);
  pack_weight_kernel<<<(wpack + 255) / 256, 256, 0, stream>>>(Wk, WBk);
  pack_weight_kernel<<<(wpack + 255) / 256, 256, 0, stream>>>(Wv, WBv);
  pack_weight_kernel<<<(wpack + 255) / 256, 256, 0, stream>>>(Wo, WBo);

  dim3 g(MROWS / 64, NH);
  gemm_bf16_wmma<<<g, 128, 0, stream>>>(xbf, WBq, qbf,  nullptr, 0);
  gemm_bf16_wmma<<<g, 128, 0, stream>>>(xbf, WBk, ktbf, nullptr, 1);
  gemm_bf16_wmma<<<g, 128, 0, stream>>>(xbf, WBv, vprj, nullptr, 2);

  dim3 ga(TSEQ / 16, NH, NB);
  flash_attn_wmma<<<ga, 32, 0, stream>>>(qbf, ktbf, vprj, shx, lam, attnbf);

  gemm_bf16_wmma<<<g, 128, 0, stream>>>(attnbf, WBo, nullptr, out, 3);
}
